// LogitsProjector_16767552323688
// MI455X (gfx1250) — matrix operand order repaired and checked by reading, not compiled, not run on hardware
//
#include <hip/hip_runtime.h>

// GEMM: C[2048,25000] = A[2048,32000] * B[25000,32000]^T  (f32 in/out, bf16 WMMA compute)
// MI455X / gfx1250: wave32, v_wmma_f32_16x16x32_bf16, TDM (tensor_load_to_lds) staging,
// double-buffered LDS, 64x64 wave tiles, rotated B-frag registers (no WMMA->VALU WAR nops),
// L2-aware block swizzle (4 M-panels resident while streaming B).

#define M_TOK 2048
#define N_OUT 25000
#define K_DIM 32000

#define TILE_M 256                 // WG M tile (4 waves x 64 rows)
#define TILE_N 128                 // WG N tile (2 waves x 64 cols)
#define TILE_K 32
#define K_STEPS (K_DIM / TILE_K)   // 1000

#define M_BLOCKS (M_TOK / TILE_M)                    // 8
#define N_BLOCKS ((N_OUT + TILE_N - 1) / TILE_N)     // 196
#define GROUP_M 4                                    // 4*32.8MB A slice fits 192MB L2

// TDM writes f32 rows with hardware padding: 32 f32 (128B) + 16B pad = 144B pitch.
// 144/4 = 36 words -> 36*r mod 64 distinct for r=0..15: conflict-free frag reads.
#define F32_PITCH   144
#define A_PLANE     (TILE_M * F32_PITCH)   // 36864
#define B_PLANE     (TILE_N * F32_PITCH)   // 18432
#define BUF_BYTES   (A_PLANE + B_PLANE)    // 55296
// total LDS: 2 buffers = 110592 B  (<= 320KB/WGP; 2 WGs per WGP still fit)

typedef __attribute__((ext_vector_type(4)))  float        f32x4;
typedef __attribute__((ext_vector_type(4)))  unsigned int u32x4;
typedef __attribute__((ext_vector_type(8)))  unsigned int u32x8;
typedef __attribute__((ext_vector_type(8)))  float        f32x8;
typedef __attribute__((ext_vector_type(16))) __bf16       bf16x16;

union Cvt16 { f32x4 v4[4]; float f[16]; };
union PackU { bf16x16 v; __bf16 h[16]; };

// ---- Tensor Data Mover: 2D f32 tile (TILE_K wide x rows) global -> LDS ----
// D# per CDNA5 ISA. Groups 2/3 omitted (2D tensor).
// OOB rows (row >= tensor_d1) are zero-filled by hardware -> handles N edge.
__device__ __forceinline__ void tdm_load_2d_f32(unsigned lds_byte_addr,
                                                const float* gsrc,
                                                unsigned tensor_d0,   // elems remaining in K
                                                unsigned tensor_d1,   // rows remaining
                                                unsigned tile_rows,
                                                unsigned stride0) {   // row stride in elems
  unsigned long long ga = (unsigned long long)(uintptr_t)gsrc;
  u32x4 g0;
  g0.x = 1u;                                                // count=1, user mode, no gather
  g0.y = lds_byte_addr;                                     // lds_addr [63:32]
  g0.z = (unsigned)ga;                                      // global_addr[31:0]
  g0.w = (unsigned)((ga >> 32) & 0x01FFFFFFu) | (2u << 30); // addr[56:32], type=2 (image)
  u32x8 g1;
  g1.s0 = (2u << 16)      // data_size = 2 -> 4 bytes
        | (1u << 20)      // pad_enable
        | (4u << 22)      // pad_interval = 4 -> every 32 DWORDs (128B row)
        | (3u << 25);     // pad_amount   = 3 -> 4 DWORDs (16B)  => 144B pitch
  g1.s1 = (tensor_d0 & 0xFFFFu) << 16;                      // tensor_dim0[15:0]
  g1.s2 = (tensor_d0 >> 16) | ((tensor_d1 & 0xFFFFu) << 16);
  g1.s3 = (tensor_d1 >> 16) | ((unsigned)TILE_K << 16);     // tile_dim0 = 32
  g1.s4 = tile_rows;                                        // tile_dim1, tile_dim2 = 0
  g1.s5 = stride0;                                          // tensor_dim0_stride[31:0]
  g1.s6 = 0u;
  g1.s7 = 0u;
  asm volatile("tensor_load_to_lds %0, %1" :: "s"(g0), "s"(g1) : "memory");
}

__device__ __forceinline__ bf16x16 cvt16(const Cvt16& r) {
  PackU pk;
#pragma unroll
  for (int e = 0; e < 16; ++e) pk.h[e] = (__bf16)r.f[e];    // -> v_cvt_pk_bf16_f32
  return pk.v;
}

__global__ __launch_bounds__(256)
void logits_proj_gemm_bf16_wmma(const float* __restrict__ A,
                                const float* __restrict__ B,
                                float* __restrict__ C) {
  __shared__ __align__(16) unsigned char ldsRaw[2 * BUF_BYTES];

  const int tid  = threadIdx.x;
  const int lane = tid & 31;
  const int wave = tid >> 5;        // 0..7
  const int wm   = wave & 3;        // 4 waves x 64 rows = 256 M
  const int wn   = wave >> 2;       // 2 waves x 64 cols = 128 N
  const int lmod  = lane & 15;
  const int lquad = lane >> 4;      // which K-half this lane holds

  // ---- L2-aware swizzle: group 4 M-panels (A slice ~131MB, L2-resident),
  //      march N across full vocab so B streams from HBM only twice total. ----
  const int bid   = blockIdx.x;
  const int group = bid / (GROUP_M * N_BLOCKS);
  const int rem   = bid % (GROUP_M * N_BLOCKS);
  const int nb    = rem / GROUP_M;
  const int mb    = group * GROUP_M + (rem % GROUP_M);
  const int mBase = mb * TILE_M;
  const int nBase = nb * TILE_N;

  const unsigned ldsBase = (unsigned)(uintptr_t)(void*)&ldsRaw[0];

  f32x8 acc[4][4];                  // 64x64 wave tile: 128 accumulator VGPRs
#pragma unroll
  for (int i = 0; i < 4; ++i)
#pragma unroll
    for (int j = 0; j < 4; ++j)
#pragma unroll
      for (int v = 0; v < 8; ++v) acc[i][j][v] = 0.0f;

  // ---- prologue: TDM-load k-tile 0 into buffer 0 (A on wave 0, B on wave 1) ----
  if (wave == 0) {
    tdm_load_2d_f32(ldsBase, A + (size_t)mBase * K_DIM,
                    (unsigned)K_DIM, (unsigned)(M_TOK - mBase),
                    (unsigned)TILE_M, (unsigned)K_DIM);
    __builtin_amdgcn_s_wait_tensorcnt(0);
  } else if (wave == 1) {
    tdm_load_2d_f32(ldsBase + A_PLANE, B + (size_t)nBase * K_DIM,
                    (unsigned)K_DIM, (unsigned)(N_OUT - nBase),
                    (unsigned)TILE_N, (unsigned)K_DIM);
    __builtin_amdgcn_s_wait_tensorcnt(0);
  }
  __syncthreads();

#pragma unroll 1
  for (int kt = 0; kt < K_STEPS; ++kt) {
    const int buf = kt & 1;

    // TDM for next tile overlaps this tile's compute (A: wave 0, B: wave 1)
    if (kt + 1 < K_STEPS) {
      const unsigned dst = ldsBase + (unsigned)(buf ^ 1) * BUF_BYTES;
      const unsigned kRem = (unsigned)(K_DIM - (kt + 1) * TILE_K);
      if (wave == 0)
        tdm_load_2d_f32(dst, A + (size_t)mBase * K_DIM + (size_t)(kt + 1) * TILE_K,
                        kRem, (unsigned)(M_TOK - mBase), (unsigned)TILE_M, (unsigned)K_DIM);
      else if (wave == 1)
        tdm_load_2d_f32(dst + A_PLANE, B + (size_t)nBase * K_DIM + (size_t)(kt + 1) * TILE_K,
                        kRem, (unsigned)(N_OUT - nBase), (unsigned)TILE_N, (unsigned)K_DIM);
    }

    const char* la = (const char*)ldsRaw + buf * BUF_BYTES;
    const char* lb = la + A_PLANE;

    // A frags (16x32 bf16): lane<16 holds K{0..15}, lane>=16 holds K{16..31}
    bf16x16 aF[4];
#pragma unroll
    for (int i = 0; i < 4; ++i) {
      const char* p = la + (wm * 64 + i * 16 + lmod) * F32_PITCH + lquad * 64;
      Cvt16 t;
      t.v4[0] = *(const f32x4*)(p);
      t.v4[1] = *(const f32x4*)(p + 16);
      t.v4[2] = *(const f32x4*)(p + 32);
      t.v4[3] = *(const f32x4*)(p + 48);
      aF[i] = cvt16(t);
    }

    // B frags (32x16 bf16), register-rotated pipeline:
    //   loads(j+1) issued early -> WMMA group j -> cvt(j+1) into a fresh SSA value.
    // Distinct live ranges => new VGPRs; the 4 WMMAs of group j provide the required
    // 4-coexec spacing after group j-1, so no WMMA->VALU WAR v_nops are needed.
    Cvt16 t0;
    {
      const char* p = lb + (wn * 64 + 0 * 16 + lmod) * F32_PITCH + lquad * 64;
      t0.v4[0] = *(const f32x4*)(p);
      t0.v4[1] = *(const f32x4*)(p + 16);
      t0.v4[2] = *(const f32x4*)(p + 32);
      t0.v4[3] = *(const f32x4*)(p + 48);
    }
    bf16x16 bFcur = cvt16(t0);
#pragma unroll
    for (int j = 0; j < 4; ++j) {
      Cvt16 tN;
      if (j + 1 < 4) {
        const char* p = lb + (wn * 64 + (j + 1) * 16 + lmod) * F32_PITCH + lquad * 64;
        tN.v4[0] = *(const f32x4*)(p);
        tN.v4[1] = *(const f32x4*)(p + 16);
        tN.v4[2] = *(const f32x4*)(p + 32);
        tN.v4[3] = *(const f32x4*)(p + 48);
      }
#pragma unroll
      for (int i = 0; i < 4; ++i)
        acc[i][j] = __builtin_amdgcn_wmma_f32_16x16x32_bf16(
            false, aF[i], false, bFcur, (short)0, acc[i][j], false, false);
      if (j + 1 < 4) bFcur = cvt16(tN);
    }

    if (wave <= 1 && kt + 1 < K_STEPS)
      __builtin_amdgcn_s_wait_tensorcnt(0);
    __syncthreads();
  }

  // ---- epilogue: C/D layout: lane 0-15 -> N=lane, VGPR v -> M=v (+8 for lanes 16-31) ----
#pragma unroll
  for (int i = 0; i < 4; ++i) {
#pragma unroll
    for (int j = 0; j < 4; ++j) {
      const int col = nBase + wn * 64 + j * 16 + lmod;
      if (col < N_OUT) {
        const int rowb = mBase + wm * 64 + i * 16 + (lquad ? 8 : 0);
#pragma unroll
        for (int v = 0; v < 8; ++v)
          C[(size_t)(rowb + v) * N_OUT + col] = acc[i][j][v];
      }
    }
  }
}

extern "C" void kernel_launch(void* const* d_in, const int* in_sizes, int n_in,
                              void* d_out, int out_size, void* d_ws, size_t ws_size,
                              hipStream_t stream) {
  (void)in_sizes; (void)n_in; (void)out_size; (void)d_ws; (void)ws_size;
  const float* A = (const float*)d_in[0];   // teacher_logits [2048, 32000]
  const float* B = (const float*)d_in[1];   // projection    [25000, 32000]
  float* C = (float*)d_out;                 // out           [2048, 25000]

  dim3 grid(M_BLOCKS * N_BLOCKS);           // 1568, swizzled in-kernel
  logits_proj_gemm_bf16_wmma<<<grid, 256, 0, stream>>>(A, B, C);
}
